// SimpleGTN_EEG_85220741087427
// MI455X (gfx1250) — compile-verified, matrix operand order checked by reading, not run on hardware
//
#include <hip/hip_runtime.h>

// ---------------------------------------------------------------------------
// CDNA5 (gfx1250) implementation of SimpleGTN_EEG forward.
// Heavy convs (conv2/conv3) = implicit GEMM on v_wmma_f32_16x16x32_f16,
// BN-apply + ReLU + maxpool fused into the LDS staging loader of the next conv.
// All cross-block reductions are fixed-order (deterministic, graph-capturable).
// Workspace use (~1.24 GiB): t1/t2/t3 f16 intermediates + small buffers.
// ---------------------------------------------------------------------------

typedef _Float16 half_t;
typedef __attribute__((ext_vector_type(16))) _Float16     v16h;
typedef __attribute__((ext_vector_type(8)))  float        v8f;
typedef __attribute__((ext_vector_type(4)))  unsigned int u32x4;

#define DEVINL static __device__ __forceinline__

DEVINL v8f zero8() { v8f z = {0.f,0.f,0.f,0.f,0.f,0.f,0.f,0.f}; return z; }

// Build a 16-half WMMA operand fragment from two 16-byte contiguous chunks.
// Per CDNA5 ISA 16-bit A/B layout: halves 0..7 = K[k0..k0+7],
// halves 8..15 = K[k0+16..k0+23], with k0 = kk*32 + (lane>=16 ? 8 : 0).
DEVINL v16h frag_ld(const half_t* p0, const half_t* p1) {
  union { v16h h; u32x4 u[2]; } f;
  f.u[0] = *(const u32x4*)p0;
  f.u[1] = *(const u32x4*)p1;
  return f.h;
}

DEVINL v8f wmma_f16(v16h a, v16h b, v8f c) {
  return __builtin_amdgcn_wmma_f32_16x16x32_f16(
      /*neg_a=*/false, a, /*neg_b=*/false, b,
      /*c_mod=*/(short)0, c, /*reuse_a=*/false, /*reuse_b=*/false);
}

// ---------------------------------------------------------------------------
// conv1: per sample (B*C = 4096), 1->64ch, k=7, pad=3, L=1000. VALU (K too
// small for WMMA). Output t1[s][pos][ch] f16 (pre-BN, needed for stats).
// ---------------------------------------------------------------------------
__global__ __launch_bounds__(256) void conv1_kernel(
    const float* __restrict__ x, const float* __restrict__ w1,
    const float* __restrict__ b1, half_t* __restrict__ t1)
{
  __shared__ float xs[1006];
  __shared__ float w1s[64*7];
  __shared__ float b1s[64];
  int s = blockIdx.x, t = threadIdx.x;
  for (int idx = t; idx < 1006; idx += 256)
    xs[idx] = (idx < 3 || idx >= 1003) ? 0.f : x[(size_t)s*1000 + idx - 3];
  for (int idx = t; idx < 448; idx += 256) w1s[idx] = w1[idx];
  if (t < 64) b1s[t] = b1[t];
  __syncthreads();
  for (int idx = t; idx < 64000; idx += 256) {
    int pos = idx >> 6, c = idx & 63;
    float acc = b1s[c];
#pragma unroll
    for (int k = 0; k < 7; ++k) acc += w1s[c*7+k] * xs[pos + k];
    t1[(size_t)s*64000 + idx] = (half_t)acc;
  }
}

// ---------------------------------------------------------------------------
// BatchNorm stats, two deterministic passes over [R rows][C ch] f16 tensors.
// ---------------------------------------------------------------------------
__global__ __launch_bounds__(256) void stats_partial_kernel(
    const half_t* __restrict__ src, int C, int rpb, float* __restrict__ part)
{
  __shared__ float rs[256], rq[256];
  int blk = blockIdx.x, t = threadIdx.x;
  int c = t % C, g = t / C, Gp = 256 / C;
  size_t r0 = (size_t)blk * rpb;
  float s = 0.f, q = 0.f;
  for (int i = g; i < rpb; i += Gp) {
    float v = (float)src[(r0 + i) * C + c];
    s += v; q += v * v;
  }
  rs[t] = s; rq[t] = q;
  __syncthreads();
  if (t < C) {               // fixed-order combine across the Gp sub-threads
    float S = 0.f, Q = 0.f;
    for (int gg = 0; gg < Gp; ++gg) { S += rs[t + gg*C]; Q += rq[t + gg*C]; }
    part[(size_t)blk*C*2 + t*2 + 0] = S;
    part[(size_t)blk*C*2 + t*2 + 1] = Q;
  }
}

__global__ __launch_bounds__(256) void stats_final_kernel(
    const float* __restrict__ part, int C, int G, float invN,
    const float* __restrict__ gamma, const float* __restrict__ beta,
    float* __restrict__ ss)
{
  __shared__ float rs[256], rq[256];
  int c = blockIdx.x, t = threadIdx.x;
  float s = 0.f, q = 0.f;
  for (int i = t; i < G; i += 256) {
    s += part[(size_t)i*C*2 + c*2 + 0];
    q += part[(size_t)i*C*2 + c*2 + 1];
  }
  rs[t] = s; rq[t] = q; __syncthreads();
  for (int off = 128; off > 0; off >>= 1) {
    if (t < off) { rs[t] += rs[t+off]; rq[t] += rq[t+off]; }
    __syncthreads();
  }
  if (t == 0) {
    float m = rs[0] * invN;
    float v = rq[0] * invN - m * m;
    float sc = gamma[c] / sqrtf(v + 1e-5f);
    ss[c*2+0] = sc;
    ss[c*2+1] = beta[c] - m * sc;
  }
}

// ---------------------------------------------------------------------------
// Weight/bias prep: f16 weights in tap-major K order (K = tap*Cin + c) and
// transposed GTN weights Wt[e][o][c]; mean-over-edges biases.
// ---------------------------------------------------------------------------
__global__ __launch_bounds__(256) void prep_kernel(
    const float* __restrict__ c2w, const float* __restrict__ c3w,
    const float* __restrict__ g1w, const float* __restrict__ g1b,
    const float* __restrict__ g2w, const float* __restrict__ g2b,
    half_t* __restrict__ w2f, half_t* __restrict__ w3f,
    half_t* __restrict__ W1t, half_t* __restrict__ W2t,
    float* __restrict__ bb1, float* __restrict__ bb2)
{
  int t = threadIdx.x;
  for (int i = t; i < 128*320; i += 256) {     // w2f[o][tap*64+c]
    int o = i / 320, rem = i % 320, tap = rem >> 6, c = rem & 63;
    w2f[i] = (half_t)c2w[(o*64 + c)*5 + tap];
  }
  for (int i = t; i < 128*384; i += 256) {     // w3f[o][tap*128+c]
    int o = i / 384, rem = i % 384, tap = rem >> 7, c = rem & 127;
    w3f[i] = (half_t)c3w[(o*128 + c)*3 + tap];
  }
  for (int i = t; i < 3*64*128; i += 256) {    // W1t[e][o][c] = g1w[e][c][o]
    int e = i / 8192, rem = i & 8191, o = rem >> 7, c = rem & 127;
    W1t[i] = (half_t)g1w[(e*128 + c)*64 + o];
  }
  for (int i = t; i < 3*64*64; i += 256) {     // W2t[e][o][c] = g2w[e][c][o]
    int e = i / 4096, rem = i & 4095, o = rem >> 6, c = rem & 63;
    W2t[i] = (half_t)g2w[(e*64 + c)*64 + o];
  }
  if (t < 64) {
    bb1[t] = (g1b[t] + g1b[64+t] + g1b[128+t]) * (1.f/3.f);
    bb2[t] = (g2b[t] + g2b[64+t] + g2b[128+t]) * (1.f/3.f);
  }
}

// ---------------------------------------------------------------------------
// conv2: implicit GEMM, M=128 outch (8 waves x 16), N=16 positions/tile,
// K=320 (5 taps x 64 ch, tap-major). Loader fuses bn1+relu+maxpool2 of t1
// into LDS P[504 rows (pos+2)][64 ch] f16. 32 N-tiles cover pos 0..511
// (pos>=500 discarded; halo/overrun rows are zero/clamped).
// ---------------------------------------------------------------------------
__global__ __launch_bounds__(256) void conv2_kernel(
    const half_t* __restrict__ t1, const float* __restrict__ ss1,
    const half_t* __restrict__ w2f, const float* __restrict__ b2,
    half_t* __restrict__ t2)
{
  __shared__ __align__(16) half_t P[504*64];
  int s = blockIdx.x, t = threadIdx.x;
  { // zero halo rows {0,1,502,503}
    int g = t >> 6, c = t & 63;
    int row = (g < 2) ? g : (g + 500);
    P[row*64 + c] = (half_t)0.f;
  }
  for (int idx = t; idx < 500*64; idx += 256) {
    int u = idx >> 6, c = idx & 63;
    float sc = ss1[c*2+0], sh = ss1[c*2+1];
    size_t base = (size_t)s*64000 + (size_t)(2*u)*64 + c;
    float v0 = sc * (float)t1[base]      + sh;
    float v1 = sc * (float)t1[base + 64] + sh;
    v0 = v0 > 0.f ? v0 : 0.f;  v1 = v1 > 0.f ? v1 : 0.f;
    P[(u+2)*64 + c] = (half_t)(v0 > v1 ? v0 : v1);
  }
  __syncthreads();

  int wave = t >> 5, lane = t & 31, ll = lane & 15, lh = lane >> 4;
  float bias[8];
#pragma unroll
  for (int r = 0; r < 8; ++r) bias[r] = b2[wave*16 + lh*8 + r];
  const half_t* wrow = w2f + (wave*16 + ll) * 320;

  for (int nt = 0; nt < 32; ++nt) {
    int t0 = nt * 16;
    v8f acc = zero8();
    for (int kk = 0; kk < 10; ++kk) {
      int k0 = kk*32 + lh*8;
      v16h a = frag_ld(wrow + k0, wrow + k0 + 16);
      int K1 = k0 + 16;
      int tap0 = k0 >> 6, c0 = k0 & 63;
      int tap1 = K1 >> 6, c1 = K1 & 63;
      int r0 = t0 + ll + tap0; if (r0 > 503) r0 = 503;   // clamp -> zero row
      int r1 = t0 + ll + tap1; if (r1 > 503) r1 = 503;
      v16h b = frag_ld(&P[r0*64 + c0], &P[r1*64 + c1]);
      acc = wmma_f16(a, b, acc);
    }
    int pos = t0 + ll;
    if (pos < 500) {
      union { u32x4 u; half_t h[8]; } pk;
#pragma unroll
      for (int r = 0; r < 8; ++r) pk.h[r] = (half_t)(acc[r] + bias[r]);
      int chb = wave*16 + lh*8;
      *(u32x4*)&t2[((size_t)s*500 + pos)*128 + chb] = pk.u;
    }
  }
}

// ---------------------------------------------------------------------------
// conv3: same scheme. K=384 (3 taps x 128 ch). LDS P[252 rows (pos+1)][128].
// 16 N-tiles cover pos 0..255 (pos>=250 discarded).
// ---------------------------------------------------------------------------
__global__ __launch_bounds__(256) void conv3_kernel(
    const half_t* __restrict__ t2, const float* __restrict__ ss2,
    const half_t* __restrict__ w3f, const float* __restrict__ b3,
    half_t* __restrict__ t3)
{
  __shared__ __align__(16) half_t P[252*128];
  int s = blockIdx.x, t = threadIdx.x;
  { // zero halo rows {0, 251}
    int g = t >> 7, c = t & 127;
    int row = g ? 251 : 0;
    P[row*128 + c] = (half_t)0.f;
  }
  for (int idx = t; idx < 250*128; idx += 256) {
    int u = idx >> 7, c = idx & 127;
    float sc = ss2[c*2+0], sh = ss2[c*2+1];
    size_t base = (size_t)s*64000 + (size_t)(2*u)*128 + c;   // 500*128 = 64000
    float v0 = sc * (float)t2[base]       + sh;
    float v1 = sc * (float)t2[base + 128] + sh;
    v0 = v0 > 0.f ? v0 : 0.f;  v1 = v1 > 0.f ? v1 : 0.f;
    P[(u+1)*128 + c] = (half_t)(v0 > v1 ? v0 : v1);
  }
  __syncthreads();

  int wave = t >> 5, lane = t & 31, ll = lane & 15, lh = lane >> 4;
  float bias[8];
#pragma unroll
  for (int r = 0; r < 8; ++r) bias[r] = b3[wave*16 + lh*8 + r];
  const half_t* wrow = w3f + (wave*16 + ll) * 384;

  for (int nt = 0; nt < 16; ++nt) {
    int t0 = nt * 16;
    v8f acc = zero8();
    for (int kk = 0; kk < 12; ++kk) {
      int k0 = kk*32 + lh*8;
      v16h a = frag_ld(wrow + k0, wrow + k0 + 16);
      int K1 = k0 + 16;
      int tap0 = k0 >> 7, c0 = k0 & 127;
      int tap1 = K1 >> 7, c1 = K1 & 127;
      int r0 = t0 + ll + tap0; if (r0 > 251) r0 = 251;
      int r1 = t0 + ll + tap1; if (r1 > 251) r1 = 251;
      v16h b = frag_ld(&P[r0*128 + c0], &P[r1*128 + c1]);
      acc = wmma_f16(a, b, acc);
    }
    int pos = t0 + ll;
    if (pos < 250) {
      union { u32x4 u; half_t h[8]; } pk;
#pragma unroll
      for (int r = 0; r < 8; ++r) pk.h[r] = (half_t)(acc[r] + bias[r]);
      int chb = wave*16 + lh*8;
      *(u32x4*)&t3[((size_t)s*250 + pos)*128 + chb] = pk.u;
    }
  }
}

// ---------------------------------------------------------------------------
// feats: X[s][c] = mean_u relu(bn3(t3[s][u][c]))  (AdaptiveAvgPool1d(1))
// ---------------------------------------------------------------------------
__global__ __launch_bounds__(256) void feats_kernel(
    const half_t* __restrict__ t3, const float* __restrict__ ss3,
    float* __restrict__ X)
{
  __shared__ float red[256];
  int s = blockIdx.x, t = threadIdx.x;
  int c = t & 127, h = t >> 7;
  float sc = ss3[c*2+0], sh = ss3[c*2+1];
  float acc = 0.f;
  for (int u = h; u < 250; u += 2) {
    float v = sc * (float)t3[((size_t)s*250 + u)*128 + c] + sh;
    acc += v > 0.f ? v : 0.f;
  }
  red[t] = acc; __syncthreads();
  if (t < 128) X[(size_t)s*128 + c] = (red[t] + red[t + 128]) * (1.f/250.f);
}

// ---------------------------------------------------------------------------
// Per-sample channel covariance of the raw input (for correlation graph).
// ---------------------------------------------------------------------------
__global__ __launch_bounds__(256) void corr_kernel(
    const float* __restrict__ x, float* __restrict__ cov)
{
  __shared__ float mean[64];
  __shared__ float buf[64*125];
  __shared__ float red[256];
  int b = blockIdx.x, t = threadIdx.x;
  { int c = t >> 2, sub = t & 3;
    float s = 0.f;
    for (int j = sub; j < 1000; j += 4) s += x[((size_t)b*64 + c)*1000 + j];
    red[t] = s; }
  __syncthreads();
  if (t < 64)
    mean[t] = (red[t*4] + red[t*4+1] + red[t*4+2] + red[t*4+3]) * (1.f/1000.f);
  __syncthreads();
  float acc[16];
#pragma unroll
  for (int i = 0; i < 16; ++i) acc[i] = 0.f;
  for (int t0 = 0; t0 < 1000; t0 += 125) {
    for (int idx = t; idx < 64*125; idx += 256) {
      int c = idx / 125;
      buf[idx] = x[((size_t)b*64 + c)*1000 + t0 + (idx % 125)] - mean[c];
    }
    __syncthreads();
    for (int i = 0; i < 16; ++i) {
      int p = t + i*256; int c = p >> 6, d = p & 63;
      float a = 0.f;
      for (int j = 0; j < 125; ++j) a += buf[c*125 + j] * buf[d*125 + j];
      acc[i] += a;
    }
    __syncthreads();
  }
  for (int i = 0; i < 16; ++i)
    cov[(size_t)b*4096 + t + i*256] = acc[i];
}

// ---------------------------------------------------------------------------
// Average correlation -> adjacency -> row-normalized A stack, emitted as f16
// A[3][64][64] for the WMMA GTN layers. Single block, fixed-order sums.
// ---------------------------------------------------------------------------
__global__ __launch_bounds__(256) void build_A_kernel(
    const float* __restrict__ cov, half_t* __restrict__ Af)
{
  __shared__ float adj[4096];
  __shared__ float rsum[64];
  int t = threadIdx.x;
  for (int i = 0; i < 16; ++i) {
    int p = t + i*256; int c = p >> 6, d = p & 63;
    float a = 0.f;
    for (int b = 0; b < 64; ++b) {
      const float* cb = cov + (size_t)b*4096;
      float den = sqrtf(cb[c*64 + c]) * sqrtf(cb[d*64 + d]);
      a += cb[c*64 + d] / den;
    }
    a *= (1.f/64.f);
    adj[p] = (fabsf(a) > 0.3f ? 1.f : 0.f) + (c == d ? 1.f : 0.f);
  }
  __syncthreads();
  if (t < 64) {
    float s = 0.f;
    for (int d = 0; d < 64; ++d) s += adj[t*64 + d];
    rsum[t] = s + 1e-6f;
  }
  __syncthreads();
  for (int i = 0; i < 16; ++i) {
    int p = t + i*256; int c = p >> 6, d = p & 63;
    Af[p]        = (half_t)(adj[p] / rsum[c]);
    Af[4096 + p] = (half_t)(c == d ? 1.f : 0.f);
    Af[8192 + p] = (half_t)(1.f/64.f);
  }
}

// ---------------------------------------------------------------------------
// Graph transform layer: out[b,n,o] = relu( (1/3) sum_e (A_e X_b W_e)[n,o]
//                                           + mean_e bias_e[o] )
// Per-b block, 8 waves. For each e: step1 T = A_e*X (WMMA, T->LDS),
// step2 acc += T*W_e (WMMA, acc in registers across e). Cin in {128,64}.
// ---------------------------------------------------------------------------
__global__ __launch_bounds__(256) void gtl_kernel(
    const float* __restrict__ Xin, int Cin,
    const half_t* __restrict__ Af, const half_t* __restrict__ Wt,
    const float* __restrict__ bbar, float* __restrict__ Xout)
{
  __shared__ __align__(16) half_t Xt[128*64];   // X^T : [c][m]
  __shared__ __align__(16) half_t T[64*128];    // T_e : [n][c]
  int b = blockIdx.x, t = threadIdx.x;
  int wave = t >> 5, lane = t & 31, ll = lane & 15, lh = lane >> 4;

  for (int idx = t; idx < Cin*64; idx += 256) {
    int c = idx >> 6, m = idx & 63;
    Xt[c*64 + m] = (half_t)Xin[((size_t)b*64 + m)*Cin + c];
  }
  __syncthreads();

  int ntC = Cin >> 4, ktC = Cin >> 5;
  v8f acc2[2] = { zero8(), zero8() };

  for (int e = 0; e < 3; ++e) {
    // step1: T = A_e (64x64) * X (64xCin)
    int tot = 4 * ntC;
    for (int tile = wave; tile < tot; tile += 8) {
      int mt = tile / ntC, nt = tile % ntC;
      v8f acc = zero8();
      for (int kk = 0; kk < 2; ++kk) {
        int k0 = kk*32 + lh*8;
        const half_t* ar = Af + (e*64 + mt*16 + ll) * 64;
        v16h a = frag_ld(ar + k0, ar + k0 + 16);
        const half_t* br = Xt + (nt*16 + ll) * 64;
        v16h bf = frag_ld(br + k0, br + k0 + 16);
        acc = wmma_f16(a, bf, acc);
      }
      int c = nt*16 + ll;
#pragma unroll
      for (int r = 0; r < 8; ++r) {
        int n = mt*16 + r + lh*8;
        T[n*Cin + c] = (half_t)acc[r];
      }
    }
    __syncthreads();
    // step2: acc += T (64xCin) * W_e (Cinx64), W stored transposed [o][c]
#pragma unroll
    for (int q = 0; q < 2; ++q) {
      int tile = wave + q*8; int mt = tile >> 2, nt = tile & 3;
      v8f a2 = acc2[q];
      for (int kk = 0; kk < ktC; ++kk) {
        int k0 = kk*32 + lh*8;
        const half_t* ar = T + (mt*16 + ll) * Cin;
        v16h af = frag_ld(ar + k0, ar + k0 + 16);
        const half_t* br = Wt + (e*64 + nt*16 + ll) * Cin;
        v16h bf = frag_ld(br + k0, br + k0 + 16);
        a2 = wmma_f16(af, bf, a2);
      }
      acc2[q] = a2;
    }
    __syncthreads();   // T reused next e
  }

#pragma unroll
  for (int q = 0; q < 2; ++q) {
    int tile = wave + q*8; int mt = tile >> 2, nt = tile & 3;
    int o = nt*16 + ll;
#pragma unroll
    for (int r = 0; r < 8; ++r) {
      int n = mt*16 + r + lh*8;
      float v = acc2[q][r] * (1.f/3.f) + bbar[o];
      Xout[((size_t)b*64 + n)*64 + o] = v > 0.f ? v : 0.f;
    }
  }
}

// ---------------------------------------------------------------------------
// Head: node-mean pool -> fc1+relu -> fc2. One block per batch sample.
// ---------------------------------------------------------------------------
__global__ __launch_bounds__(128) void head_kernel(
    const float* __restrict__ X2, const float* __restrict__ fc1w,
    const float* __restrict__ fc1b, const float* __restrict__ fc2w,
    const float* __restrict__ fc2b, float* __restrict__ out)
{
  __shared__ float pooled[64];
  __shared__ float h1[128];
  int b = blockIdx.x, t = threadIdx.x;
  if (t < 64) {
    float s = 0.f;
    for (int n = 0; n < 64; ++n) s += X2[((size_t)b*64 + n)*64 + t];
    pooled[t] = s * (1.f/64.f);
  }
  __syncthreads();
  { float s = fc1b[t];
    for (int k = 0; k < 64; ++k) s += pooled[k] * fc1w[t*64 + k];
    h1[t] = s > 0.f ? s : 0.f; }
  __syncthreads();
  if (t < 3) {
    float s = fc2b[t];
    for (int k = 0; k < 128; ++k) s += h1[k] * fc2w[t*128 + k];
    out[b*3 + t] = s;
  }
}

// ---------------------------------------------------------------------------
extern "C" void kernel_launch(void* const* d_in, const int* in_sizes, int n_in,
                              void* d_out, int out_size, void* d_ws, size_t ws_size,
                              hipStream_t stream) {
  (void)in_sizes; (void)n_in; (void)out_size; (void)ws_size;
  const float* x    = (const float*)d_in[0];
  const float* c1w  = (const float*)d_in[1];
  const float* c1b  = (const float*)d_in[2];
  const float* bn1g = (const float*)d_in[3];
  const float* bn1b = (const float*)d_in[4];
  const float* c2w  = (const float*)d_in[5];
  const float* c2b  = (const float*)d_in[6];
  const float* bn2g = (const float*)d_in[7];
  const float* bn2b = (const float*)d_in[8];
  const float* c3w  = (const float*)d_in[9];
  const float* c3b  = (const float*)d_in[10];
  const float* bn3g = (const float*)d_in[11];
  const float* bn3b = (const float*)d_in[12];
  const float* g1w  = (const float*)d_in[13];
  const float* g1b  = (const float*)d_in[14];
  const float* g2w  = (const float*)d_in[15];
  const float* g2b  = (const float*)d_in[16];
  const float* fc1w = (const float*)d_in[17];
  const float* fc1b = (const float*)d_in[18];
  const float* fc2w = (const float*)d_in[19];
  const float* fc2b = (const float*)d_in[20];
  float* out = (float*)d_out;

  char* ws = (char*)d_ws;
  size_t off = 0;
  auto take = [&](size_t bytes) -> char* {
    char* p = ws + off;
    off = (off + bytes + 255) & ~(size_t)255;
    return p;
  };
  half_t* t1   = (half_t*)take(524288000ULL);  // [4096][1000][64] f16
  half_t* t2   = (half_t*)take(524288000ULL);  // [4096][500][128] f16
  half_t* t3   = (half_t*)take(262144000ULL);  // [4096][250][128] f16
  float*  part = (float*)take(4194304ULL);     // [4096][128][2] stat partials
  float*  ss1  = (float*)take(512);
  float*  ss2  = (float*)take(1024);
  float*  ss3  = (float*)take(1024);
  half_t* w2f  = (half_t*)take(81920);         // [128][320] f16
  half_t* w3f  = (half_t*)take(98304);         // [128][384] f16
  float*  X0   = (float*)take(2097152);        // [64][64][128]
  float*  cov  = (float*)take(1048576);        // [64][64][64]
  half_t* Af   = (half_t*)take(24576);         // [3][64][64] f16
  half_t* W1t  = (half_t*)take(49152);         // [3][64][128] f16
  half_t* W2t  = (half_t*)take(24576);         // [3][64][64] f16
  float*  bb1  = (float*)take(256);
  float*  bb2  = (float*)take(256);
  float*  X1   = (float*)take(1048576);        // [64][64][64]
  float*  X2   = (float*)take(1048576);        // [64][64][64]

  prep_kernel<<<1, 256, 0, stream>>>(c2w, c3w, g1w, g1b, g2w, g2b,
                                     w2f, w3f, W1t, W2t, bb1, bb2);
  conv1_kernel<<<4096, 256, 0, stream>>>(x, c1w, c1b, t1);
  stats_partial_kernel<<<4096, 256, 0, stream>>>(t1, 64, 1000, part);
  stats_final_kernel<<<64, 256, 0, stream>>>(part, 64, 4096, 1.f/4096000.f,
                                             bn1g, bn1b, ss1);
  conv2_kernel<<<4096, 256, 0, stream>>>(t1, ss1, w2f, c2b, t2);
  stats_partial_kernel<<<4096, 256, 0, stream>>>(t2, 128, 500, part);
  stats_final_kernel<<<128, 256, 0, stream>>>(part, 128, 4096, 1.f/2048000.f,
                                              bn2g, bn2b, ss2);
  conv3_kernel<<<4096, 256, 0, stream>>>(t2, ss2, w3f, c3b, t3);
  stats_partial_kernel<<<4096, 256, 0, stream>>>(t3, 128, 250, part);
  stats_final_kernel<<<128, 256, 0, stream>>>(part, 128, 4096, 1.f/1024000.f,
                                              bn3g, bn3b, ss3);
  feats_kernel<<<4096, 256, 0, stream>>>(t3, ss3, X0);
  corr_kernel<<<64, 256, 0, stream>>>(x, cov);
  build_A_kernel<<<1, 256, 0, stream>>>(cov, Af);
  gtl_kernel<<<64, 256, 0, stream>>>(X0, 128, Af, W1t, bb1, X1);
  gtl_kernel<<<64, 256, 0, stream>>>(X1, 64, Af, W2t, bb2, X2);
  head_kernel<<<64, 128, 0, stream>>>(X2, fc1w, fc1b, fc2w, fc2b, out);
}
